// DeepISP_36601711296602
// MI455X (gfx1250) — matrix operand order repaired and checked by reading, not compiled, not run on hardware
//
#include <hip/hip_runtime.h>
#include <hip/hip_bf16.h>
#include <math.h>

typedef __attribute__((ext_vector_type(16))) _Float16 v16h;
typedef __attribute__((ext_vector_type(8)))  _Float16 v8h;
typedef __attribute__((ext_vector_type(8)))  float    v8f;

#define IMG  512
#define HW   (IMG * IMG)
#define KTOT 549          // 61 in-channels * 9 taps (raw OIHW layout)
#define KPAD 576          // 9 taps * 64 padded channels  (packed layout)
#define CPAD 72           // channel-minor LDS stride (bank-conflict-free)

__device__ __forceinline__ int clampi(int v, int lo, int hi) {
    return v < lo ? lo : (v > hi ? hi : v);
}

// ---------------------------------------------------------------------------
// Pre-pack conv weights: OIHW f32 [Mw x 61 x 3 x 3]  ->  f16 [64 x 576]
// with K reordered as k' = tap*64 + c (channel padded to 64, rows to 64).
// ---------------------------------------------------------------------------
__global__ __launch_bounds__(256)
void pack_weights(const float* __restrict__ w, _Float16* __restrict__ wp, int Mw)
{
    int idx = blockIdx.x * 256 + threadIdx.x;          // 64*576
    if (idx >= 64 * KPAD) return;
    int m   = idx / KPAD;
    int k   = idx - m * KPAD;
    int tap = k >> 6;
    int c   = k & 63;
    float v = (m < Mw && c < 61) ? w[m * KTOT + c * 9 + tap] : 0.f;
    wp[idx] = (_Float16)v;
}

// ---------------------------------------------------------------------------
// low0: direct conv 3->64, 3x3, pad 1 (branchless clamp+mask)
// ---------------------------------------------------------------------------
__global__ __launch_bounds__(256)
void conv_low0(const float* __restrict__ x, const float* __restrict__ w,
               const float* __restrict__ b, float* __restrict__ out)
{
    int idx = blockIdx.x * 256 + threadIdx.x;          // 64*HW threads
    if (idx >= 64 * HW) return;
    int m = idx / HW;
    int p = idx - m * HW;
    int y = p >> 9, xx = p & 511;
    float acc = b[m];
    #pragma unroll
    for (int c = 0; c < 3; ++c) {
        #pragma unroll
        for (int t = 0; t < 9; ++t) {
            int yy = y + t / 3 - 1;
            int xc = xx + t % 3 - 1;
            float msk = (yy >= 0 && yy < IMG && xc >= 0 && xc < IMG) ? 1.f : 0.f;
            float v = x[c * HW + clampi(yy, 0, IMG - 1) * IMG + clampi(xc, 0, IMG - 1)];
            acc += msk * w[m * 27 + c * 9 + t] * v;
        }
    }
    out[idx] = acc;
}

// ---------------------------------------------------------------------------
// Implicit-GEMM conv via V_WMMA_F32_16X16X32_F16 (fully unrolled K=576).
//   Per block: 64 out-channels x 32 pixels of one image row.
//   - Patch staged ONCE into channel-minor LDS: P2[3][36][72] (c 61..63 = 0),
//     global reads coalesced (c-major), LDS stores scattered.
//   - B fragments: two contiguous ds_load_b128 from P2 (no tile, no guards).
//   - A fragments: two contiguous global_load_b128 from packed f16 weights
//     (72 KB/layer -> L2-resident on 192MB L2; no LDS tile needed).
//   - K-loop contains NO barriers and NO divergence.
// ---------------------------------------------------------------------------
__global__ __launch_bounds__(256)
void conv61_wmma(const float* __restrict__ in, const _Float16* __restrict__ wp,
                 const float* __restrict__ bias, float* __restrict__ out,
                 int storeC, int do_relu)
{
    __shared__ __attribute__((aligned(32))) _Float16 P2[3][36][CPAD];

    const int tid  = threadIdx.x;
    const int lane = tid & 31;
    const int wave = tid >> 5;
    const int mb   = (wave & 3) * 16;          // channel-tile base
    const int pb   = (wave >> 2) * 16;         // pixel-tile base (within 32)
    const int y     = blockIdx.y;
    const int xbase = blockIdx.x * 32;

    // ---- zero-fill pad channels 61..63 at every (r,t)
    for (int i = tid; i < 108; i += 256) {     // 108 = 3*36
        int r = i / 36, t = i - r * 36;
        P2[r][t][61] = (_Float16)0.f;
        P2[r][t][62] = (_Float16)0.f;
        P2[r][t][63] = (_Float16)0.f;
    }
    // ---- stage receptive field (global-coalesced loads, transposed stores)
    //      P2[r][t][c] = in[c][y+r-1][xbase+t-1]  (zero outside image)
    for (int i = tid; i < 61 * 108; i += 256) {
        int c  = i / 108;
        int rt = i - c * 108;
        int r  = rt / 36, t = rt - r * 36;
        int yy = y + r - 1;
        int xx = xbase + t - 1;
        float msk = (yy >= 0 && yy < IMG && xx >= 0 && xx < IMG) ? 1.f : 0.f;
        float v = in[c * HW + clampi(yy, 0, IMG - 1) * IMG + clampi(xx, 0, IMG - 1)];
        P2[r][t][c] = (_Float16)(msk * v);
    }
    __syncthreads();

    // ---- per-lane fragment source pointers (constant across K loop)
    const int mrow  = mb + (lane & 15);
    const _Float16* wrow = wp + mrow * KPAD + ((lane < 16) ? 0 : 8);
    const int prow  = pb + (lane & 15);
    const int cfrag = (lane < 16) ? 0 : 8;

    __builtin_prefetch(wrow, 0, 1);            // global_prefetch_b8 (L2 warm)

    v8f acc = {};
    #pragma unroll
    for (int kk = 0; kk < KPAD; kk += 32) {
        const int tap = kk >> 6;               // constants after unroll
        const int ch0 = kk & 32;
        const int dy  = tap / 3;
        const int dx  = tap % 3;

        // A fragment straight from L2-resident packed weights
        v8h alo = *(const v8h*)(wrow + kk);
        v8h ahi = *(const v8h*)(wrow + kk + 16);
        v16h a;
        #pragma unroll
        for (int i = 0; i < 8; ++i) { a[i] = alo[i]; a[i + 8] = ahi[i]; }

        // B fragment straight from channel-minor patch LDS
        const _Float16* bsrc = &P2[dy][prow + dx][ch0 + cfrag];
        v8h blo = *(const v8h*)bsrc;
        v8h bhi = *(const v8h*)(bsrc + 16);
        v16h bfr;
        #pragma unroll
        for (int i = 0; i < 8; ++i) { bfr[i] = blo[i]; bfr[i + 8] = bhi[i]; }

        acc = __builtin_amdgcn_wmma_f32_16x16x32_f16(
                  false, a, false, bfr, (short)0, acc, false, false);
    }

    // ---- store D: VGPR r holds M = r (lane<16) or 8+r, N = lane&15
    int p = pb + (lane & 15);
    int x = xbase + p;
    #pragma unroll
    for (int r = 0; r < 8; ++r) {
        int m = mb + r + ((lane < 16) ? 0 : 8);
        if (m < storeC) {
            float v = acc[r] + bias[m];
            if (do_relu) v = fmaxf(v, 0.f);
            out[m * HW + y * IMG + x] = v;
        }
    }
}

// ---------------------------------------------------------------------------
// lh branch: 3->3 conv, tanh; channels 61..63 of the 64-plane buffer
// ---------------------------------------------------------------------------
__global__ __launch_bounds__(256)
void conv3_tanh(const float* __restrict__ in, const float* __restrict__ w,
                const float* __restrict__ b, float* __restrict__ out)
{
    int idx = blockIdx.x * 256 + threadIdx.x;          // 3*HW threads
    if (idx >= 3 * HW) return;
    int m = idx / HW;
    int p = idx - m * HW;
    int y = p >> 9, xx = p & 511;
    float acc = b[m];
    #pragma unroll
    for (int c = 0; c < 3; ++c) {
        #pragma unroll
        for (int t = 0; t < 9; ++t) {
            int yy = y + t / 3 - 1;
            int xc = xx + t % 3 - 1;
            float msk = (yy >= 0 && yy < IMG && xc >= 0 && xc < IMG) ? 1.f : 0.f;
            float v = in[(61 + c) * HW + clampi(yy, 0, IMG - 1) * IMG + clampi(xc, 0, IMG - 1)];
            acc += msk * w[m * 27 + c * 9 + t] * v;
        }
    }
    out[(61 + m) * HW + p] = tanhf(acc);
}

// ---------------------------------------------------------------------------
// relu(conv 64->64 stride 2) fused with 2x2 maxpool. H -> Hp = H/4.
// ---------------------------------------------------------------------------
__global__ __launch_bounds__(256)
void conv_s2_pool(const float* __restrict__ in, int H,
                  const float* __restrict__ w, const float* __restrict__ b,
                  float* __restrict__ out, int Hp)
{
    int idx = blockIdx.x * 256 + threadIdx.x;          // 64*Hp*Hp threads
    int tot = 64 * Hp * Hp;
    if (idx >= tot) return;
    int m  = idx / (Hp * Hp);
    int pp = idx - m * Hp * Hp;
    int py = pp / Hp, px = pp - py * Hp;

    float best = -1e30f;
    #pragma unroll
    for (int i = 0; i < 2; ++i) {
        #pragma unroll
        for (int j = 0; j < 2; ++j) {
            int cy = 2 * py + i;                       // conv-output coords
            int cx = 2 * px + j;
            float acc = b[m];
            for (int c = 0; c < 64; ++c) {
                #pragma unroll
                for (int t = 0; t < 9; ++t) {
                    int yy = 2 * cy + t / 3 - 1;
                    int xx = 2 * cx + t % 3 - 1;
                    float msk = (yy >= 0 && yy < H && xx >= 0 && xx < H) ? 1.f : 0.f;
                    float v = in[c * H * H + clampi(yy, 0, H - 1) * H + clampi(xx, 0, H - 1)];
                    acc += msk * w[m * 576 + c * 9 + t] * v;
                }
            }
            acc  = fmaxf(acc, 0.f);
            best = fmaxf(best, acc);
        }
    }
    out[m * Hp * Hp + pp] = best;
}

// ---------------------------------------------------------------------------
// global mean over 8x8 + linear head -> Wm[30]
// ---------------------------------------------------------------------------
__global__ void head_kernel(const float* __restrict__ h3,
                            const float* __restrict__ lw,
                            const float* __restrict__ lb,
                            float* __restrict__ Wm)
{
    __shared__ float g[64];
    int t = threadIdx.x;                                // 64 threads
    float s = 0.f;
    for (int i = 0; i < 64; ++i) s += h3[t * 64 + i];
    g[t] = s * (1.f / 64.f);
    __syncthreads();
    if (t < 30) {
        float a = lb[t];
        for (int c = 0; c < 64; ++c) a += lw[t * 64 + c] * g[c];
        Wm[t] = a;
    }
}

// ---------------------------------------------------------------------------
// final per-pixel polynomial color transform (bandwidth-bound tail)
// ---------------------------------------------------------------------------
__global__ __launch_bounds__(256)
void poly_apply(const float* __restrict__ I, const float* __restrict__ Wmg,
                float* __restrict__ out)
{
    __shared__ float wm[30];
    if (threadIdx.x < 30) wm[threadIdx.x] = Wmg[threadIdx.x];
    __syncthreads();
    int p = blockIdx.x * 256 + threadIdx.x;
    if (p >= HW) return;
    float r = I[61 * HW + p];
    float g = I[62 * HW + p];
    float c = I[63 * HW + p];
    float f[10] = { r * r, r * g, r * c, r, g * g, g * c, g, c * c, c, 1.f };
    #pragma unroll
    for (int o = 0; o < 3; ++o) {
        float a = 0.f;
        #pragma unroll
        for (int k = 0; k < 10; ++k) a += wm[o * 10 + k] * f[k];
        out[o * HW + p] = a;
    }
}

// ---------------------------------------------------------------------------
extern "C" void kernel_launch(void* const* d_in, const int* in_sizes, int n_in,
                              void* d_out, int out_size, void* d_ws, size_t ws_size,
                              hipStream_t stream)
{
    const float* x       = (const float*)d_in[0];
    const float* low0_w  = (const float*)d_in[1];
    const float* low0_b  = (const float*)d_in[2];
    const float* ll_rh_w = (const float*)d_in[3];   // (4,61,61,3,3)
    const float* ll_rh_b = (const float*)d_in[4];   // (4,61)
    const float* ll_lh_w = (const float*)d_in[5];   // (4,3,3,3,3)
    const float* ll_lh_b = (const float*)d_in[6];   // (4,3)
    const float* hl0_w   = (const float*)d_in[7];   // (64,61,3,3)
    const float* hl0_b   = (const float*)d_in[8];
    const float* hl_w    = (const float*)d_in[9];   // (3,64,64,3,3)
    const float* hl_b    = (const float*)d_in[10];  // (3,64)
    const float* lin_w   = (const float*)d_in[11];  // (30,64)
    const float* lin_b   = (const float*)d_in[12];  // (30,)

    float* ws   = (float*)d_ws;
    float* bufA = ws;                                // 64*HW f32
    float* bufB = bufA + 64 * (size_t)HW;            // 64*HW f32
    float* h1   = bufB + 64 * (size_t)HW;            // 64*128*128
    float* h2   = h1 + 64 * 128 * 128;               // 64*32*32
    float* h3   = h2 + 64 * 32 * 32;                 // 64*8*8
    float* Wm   = h3 + 64 * 64;                      // 32 (16B aligned)
    _Float16* wpack = (_Float16*)(Wm + 32);          // 5 * 64*576 f16

    // ---- pre-pack the 5 WMMA conv layers' weights (tiny, L2-resident)
    const int wpgrid = (64 * KPAD) / 256;
    for (int i = 0; i < 4; ++i)
        pack_weights<<<wpgrid, 256, 0, stream>>>(
            ll_rh_w + (size_t)i * 61 * KTOT, wpack + (size_t)i * 64 * KPAD, 61);
    pack_weights<<<wpgrid, 256, 0, stream>>>(hl0_w, wpack + (size_t)4 * 64 * KPAD, 64);

    // I = conv(x, low0)  -> bufA (64 planes)
    conv_low0<<<(64 * HW) / 256, 256, 0, stream>>>(x, low0_w, low0_b, bufA);

    // 4x { rh = relu(conv61), lh = tanh(conv3) } ping-pong bufA <-> bufB
    float* cur = bufA;
    float* nxt = bufB;
    for (int i = 0; i < 4; ++i) {
        conv61_wmma<<<dim3(IMG / 32, IMG), 256, 0, stream>>>(
            cur, wpack + (size_t)i * 64 * KPAD, ll_rh_b + i * 61, nxt,
            /*storeC=*/61, /*relu=*/1);
        conv3_tanh<<<(3 * HW) / 256, 256, 0, stream>>>(
            cur, ll_lh_w + i * 81, ll_lh_b + i * 3, nxt);
        float* t = cur; cur = nxt; nxt = t;
    }
    // after 4 iterations: cur == bufA holds final I

    // h = conv(I[:,:61], hl0)  (no activation) -> bufB
    conv61_wmma<<<dim3(IMG / 32, IMG), 256, 0, stream>>>(
        cur, wpack + (size_t)4 * 64 * KPAD, hl0_b, nxt, /*storeC=*/64, /*relu=*/0);

    // 3x { relu(conv s2) + maxpool 2x2 }: 512 -> 128 -> 32 -> 8
    conv_s2_pool<<<(64 * 128 * 128) / 256, 256, 0, stream>>>(
        nxt, 512, hl_w + 0 * 64 * 576, hl_b + 0, h1, 128);
    conv_s2_pool<<<(64 * 32 * 32) / 256, 256, 0, stream>>>(
        h1, 128, hl_w + 1 * 64 * 576, hl_b + 64, h2, 32);
    conv_s2_pool<<<(64 * 8 * 8 + 255) / 256, 256, 0, stream>>>(
        h2, 32, hl_w + 2 * 64 * 576, hl_b + 128, h3, 8);

    // mean + linear -> Wm[30]
    head_kernel<<<1, 64, 0, stream>>>(h3, lin_w, lin_b, Wm);

    // final polynomial color transform from I[:,61:] (still intact in bufA)
    poly_apply<<<HW / 256, 256, 0, stream>>>(cur, Wm, (float*)d_out);
}
